// DICE_3564822855759
// MI455X (gfx1250) — matrix operand order repaired and checked by reading, not compiled, not run on hardware
//
#include <hip/hip_runtime.h>
#include <hip/hip_bf16.h>
#include <math.h>

// ---------------- problem constants (match reference) ----------------
constexpr int NN   = 50000;    // nodes
constexpr int NE   = 600000;   // edges
constexpr int NG   = 512;      // graphs
constexpr int HD   = 128;      // hidden dim
constexpr int NFIN = 9;
constexpr int EFIN = 5;
constexpr int FOUT = 64;       // NF_OUT == EF_OUT
constexpr int NCLS = 9;        // node classes
constexpr int ECLS = 5;        // edge classes

typedef __attribute__((ext_vector_type(16))) __bf16 v16bf;
typedef __attribute__((ext_vector_type(8)))  float  v8f;
typedef __attribute__((ext_vector_type(4)))  float  v4f;

// ---------------- helpers ----------------
__device__ __forceinline__ __bf16 f2bf(float f) {
#if defined(__gfx1250__)
  return (__bf16)f;                     // lowers to v_cvt_pk_bf16_f32 pairs
#else
  unsigned u = __float_as_uint(f);
  unsigned r = (u + 0x7FFFu + ((u >> 16) & 1u)) >> 16;   // RNE
  unsigned short h = (unsigned short)r;
  return __builtin_bit_cast(__bf16, h);
#endif
}

__device__ __forceinline__ void atomicMaxF(float* a, float v) {
  if (v >= 0.0f) atomicMax((int*)a, __float_as_int(v));
  else           atomicMin((unsigned int*)a, (unsigned int)__float_as_int(v));
}

// ---------------- weight packing into B-fragment layout ----------------
// W is [K x Nc] fp32 row-major.  Fragment layout for v_wmma_f32_16x16x32_bf16:
// out[((chunk*ntiles + t)*32 + lane)*16 + j] =
//   bf16( W[ (chunk*32 + (lane>=16?8:0) + (j&7) + (j>=8?16:0)) , t*16 + (lane&15) ] )   (0 if k>=K)
__global__ void pack_w(const float* __restrict__ W, __bf16* __restrict__ out,
                       int K, int Nc) {
  int ntiles  = Nc >> 4;
  int nchunks = (K + 31) >> 5;
  int total   = nchunks * ntiles * 32 * 16;
  int idx = blockIdx.x * blockDim.x + threadIdx.x;
  if (idx >= total) return;
  int j    = idx & 15;
  int lane = (idx >> 4) & 31;
  int rest = idx >> 9;
  int t    = rest % ntiles;
  int c    = rest / ntiles;
  int col  = t * 16 + (lane & 15);
  int k    = c * 32 + ((lane >> 4) << 3) + (j & 7) + ((j >> 3) << 4);
  float v  = (k < K) ? W[(size_t)k * Nc + col] : 0.0f;
  out[idx] = f2bf(v);
}

// ---------------- A-fragment loader (documented 16-bit 16x32 layout) ----
// lane<16: row = lane,     K = {kc+0..7,  kc+16..23}
// lane>=16: row = lane-16, K = {kc+8..15, kc+24..31}
template<bool K32>
__device__ __forceinline__ v16bf load_a_frag(const float* __restrict__ xr,
                                             int kbase, int K) {
  v16bf a;
  if (K32) {
    const v4f* xr4 = (const v4f*)xr;
    int q = kbase >> 2;                 // kbase is a multiple of 4
    v4f g0 = xr4[q];                    // K = kbase .. kbase+3
    v4f g1 = xr4[q + 1];                // K = kbase+4 .. kbase+7
    v4f g2 = xr4[q + 4];                // K = kbase+16 .. kbase+19
    v4f g3 = xr4[q + 5];                // K = kbase+20 .. kbase+23
#pragma unroll
    for (int j = 0; j < 4; ++j) {
      a[j]      = f2bf(g0[j]);
      a[4 + j]  = f2bf(g1[j]);
      a[8 + j]  = f2bf(g2[j]);
      a[12 + j] = f2bf(g3[j]);
    }
  } else {
#pragma unroll
    for (int j = 0; j < 8; ++j) {
      int k0 = kbase + j;
      int k1 = kbase + 16 + j;
      a[j]     = f2bf((k0 < K) ? xr[k0] : 0.0f);
      a[8 + j] = f2bf((k1 < K) ? xr[k1] : 0.0f);
    }
  }
  return a;
}

// ---------------- WMMA GEMM: Y = act(X @ W + b) ----------------
// X [M x K] fp32 row-major, Wp packed bf16 fragments, Y [M x NC] fp32.
// Single-tile kernel (16 rows per wave), used for remainder tiles.
template<int NC, int ACT, bool K32>
__global__ __launch_bounds__(128, 1)
void gemm_wmma(const float* __restrict__ X, const __bf16* __restrict__ Wp,
               const float* __restrict__ bias, float* __restrict__ Y,
               int M, int K) {
  constexpr int NT = NC / 16;
  const int lane = threadIdx.x & 31;
  const int wid  = threadIdx.x >> 5;
  const int row0 = (blockIdx.x * 4 + wid) * 16;
  if (row0 >= M) return;
  const int hi    = lane >> 4;
  const int nchunks = (K + 31) >> 5;

  v8f acc[NT] = {};
  const float* xr = X + (size_t)(row0 + (lane & 15)) * K;

  for (int c = 0; c < nchunks; ++c) {
    v16bf a = load_a_frag<K32>(xr, c * 32 + hi * 8, K);
    const __bf16* wb = Wp + ((size_t)c * NT) * 32 * 16 + (size_t)lane * 16;
#pragma unroll
    for (int t = 0; t < NT; ++t) {
      v16bf b = *(const v16bf*)(wb + (size_t)t * 32 * 16);
      acc[t] = __builtin_amdgcn_wmma_f32_16x16x32_bf16(
          false, a, false, b, (short)0, acc[t], false, false);
    }
  }
#pragma unroll
  for (int t = 0; t < NT; ++t) {
    int col  = t * 16 + (lane & 15);
    float bv = bias[col];
#pragma unroll
    for (int r = 0; r < 8; ++r) {
      float v = acc[t][r] + bv;
      if (ACT == 1) v = fmaxf(v, 0.0f);
      Y[(size_t)(row0 + r + hi * 8) * NC + col] = v;
    }
  }
}

// Two-tile kernel: each wave computes rows [32p, 32p+32); every B fragment
// feeds two WMMAs (halves B traffic, doubles WMMA:load ratio).
template<int NC, int ACT, bool K32>
__global__ __launch_bounds__(128, 1)
void gemm_wmma2(const float* __restrict__ X, const __bf16* __restrict__ Wp,
                const float* __restrict__ bias, float* __restrict__ Y,
                int M, int K) {
  constexpr int NT = NC / 16;
  const int lane = threadIdx.x & 31;
  const int wid  = threadIdx.x >> 5;
  const int row0 = (blockIdx.x * 4 + wid) * 32;
  if (row0 + 32 > M) return;
  const int hi    = lane >> 4;
  const int nchunks = (K + 31) >> 5;

  v8f accA[NT] = {}, accB[NT] = {};
  const float* xrA = X + (size_t)(row0 + (lane & 15)) * K;
  const float* xrB = xrA + (size_t)16 * K;

  for (int c = 0; c < nchunks; ++c) {
    const int kbase = c * 32 + hi * 8;
    v16bf aA = load_a_frag<K32>(xrA, kbase, K);
    v16bf aB = load_a_frag<K32>(xrB, kbase, K);
    const __bf16* wb = Wp + ((size_t)c * NT) * 32 * 16 + (size_t)lane * 16;
#pragma unroll
    for (int t = 0; t < NT; ++t) {
      v16bf b = *(const v16bf*)(wb + (size_t)t * 32 * 16);
      accA[t] = __builtin_amdgcn_wmma_f32_16x16x32_bf16(
          false, aA, false, b, (short)0, accA[t], false, false);
      accB[t] = __builtin_amdgcn_wmma_f32_16x16x32_bf16(
          false, aB, false, b, (short)0, accB[t], false, false);
    }
  }
#pragma unroll
  for (int t = 0; t < NT; ++t) {
    int col  = t * 16 + (lane & 15);
    float bv = bias[col];
#pragma unroll
    for (int r = 0; r < 8; ++r) {
      int rr = r + hi * 8;
      float vA = accA[t][r] + bv;
      float vB = accB[t][r] + bv;
      if (ACT == 1) { vA = fmaxf(vA, 0.0f); vB = fmaxf(vB, 0.0f); }
      Y[(size_t)(row0 + rr) * NC + col]      = vA;
      Y[(size_t)(row0 + 16 + rr) * NC + col] = vB;
    }
  }
}

// ---------------- BatchNorm (batch stats) + exact GELU ----------------
__global__ void col_stats(const float* __restrict__ Hm, float* __restrict__ gsum,
                          size_t rows) {
  __shared__ float ls[2 * HD];
  for (int i = threadIdx.x; i < 2 * HD; i += blockDim.x) ls[i] = 0.f;
  __syncthreads();
  size_t base   = (size_t)blockIdx.x * blockDim.x + threadIdx.x;
  size_t total4 = rows * (HD / 4);
  size_t stride = (size_t)gridDim.x * blockDim.x;     // multiple of 32
  int q = (int)(base & (HD / 4 - 1));                 // fixed column quad
  v4f s = {}, s2 = {};
  for (size_t i = base; i < total4; i += stride) {
    v4f v = ((const v4f*)Hm)[i];
    s += v;
    s2 += v * v;
  }
  float* lsp  = ls + q * 4;
  float* ls2p = ls + HD + q * 4;
  atomicAdd(lsp + 0, s[0]);  atomicAdd(lsp + 1, s[1]);
  atomicAdd(lsp + 2, s[2]);  atomicAdd(lsp + 3, s[3]);
  atomicAdd(ls2p + 0, s2[0]); atomicAdd(ls2p + 1, s2[1]);
  atomicAdd(ls2p + 2, s2[2]); atomicAdd(ls2p + 3, s2[3]);
  __syncthreads();
  if (threadIdx.x < 2 * HD) atomicAdd(&gsum[threadIdx.x], ls[threadIdx.x]);
}

__global__ void bn_finalize(const float* __restrict__ gsum, const float* __restrict__ gamma,
                            const float* __restrict__ beta, float* __restrict__ ss,
                            float invM) {
  int c = threadIdx.x;                 // 128 threads
  float mu  = gsum[c] * invM;
  float var = gsum[HD + c] * invM - mu * mu;
  float sc  = gamma[c] * rsqrtf(var + 1e-5f);
  ss[c]      = sc;
  ss[HD + c] = beta[c] - mu * sc;
}

__global__ void bn_gelu(float* __restrict__ Hm, const float* __restrict__ ss,
                        size_t total4) {
  size_t i = (size_t)blockIdx.x * blockDim.x + threadIdx.x;
  if (i >= total4) return;
  int q = (int)(i & (HD / 4 - 1));
  v4f v  = ((const v4f*)Hm)[i];
  v4f sc = ((const v4f*)ss)[q];
  v4f sh = ((const v4f*)(ss + HD))[q];
  v4f x  = v * sc + sh;
  v4f r;
#pragma unroll
  for (int k = 0; k < 4; ++k)
    r[k] = 0.5f * x[k] * (1.0f + erff(x[k] * 0.70710678118654752f));
  ((v4f*)Hm)[i] = r;
}

// ---------------- misc elementwise / graph kernels ----------------
__global__ void fill_f(float* __restrict__ p, float v, size_t n) {
  size_t i = (size_t)blockIdx.x * blockDim.x + threadIdx.x;
  if (i < n) p[i] = v;
}

__global__ void edge_attn(const float* __restrict__ n_h, const float* __restrict__ e_h,
                          const int* __restrict__ src, const int* __restrict__ dst,
                          float* __restrict__ attn) {
  int e = blockIdx.x * (blockDim.x >> 5) + (threadIdx.x >> 5);
  if (e >= NE) return;
  int lane = threadIdx.x & 31;
  v4f s  = ((const v4f*)(n_h + (size_t)src[e] * HD))[lane];
  v4f d  = ((const v4f*)(n_h + (size_t)dst[e] * HD))[lane];
  v4f ee = ((const v4f*)(e_h + (size_t)e * HD))[lane];
  float sum = (s[0] + ee[0]) * d[0] + (s[1] + ee[1]) * d[1]
            + (s[2] + ee[2]) * d[2] + (s[3] + ee[3]) * d[3];
  for (int off = 16; off > 0; off >>= 1) sum += __shfl_xor(sum, off, 32);
  if (lane == 0) attn[e] = sum;
}

__global__ void seg_max(const float* __restrict__ attn, const int* __restrict__ dst,
                        float* __restrict__ m) {
  int e = blockIdx.x * blockDim.x + threadIdx.x;
  if (e < NE) atomicMaxF(&m[dst[e]], attn[e]);
}

__global__ void edge_exp(const float* __restrict__ attn, const float* __restrict__ m,
                         const int* __restrict__ dst, float* __restrict__ exv,
                         float* __restrict__ den) {
  int e = blockIdx.x * blockDim.x + threadIdx.x;
  if (e >= NE) return;
  float ex = expf(attn[e] - m[dst[e]]);
  exv[e] = ex;
  atomicAdd(&den[dst[e]], ex);
}

// one thread per (edge, column-quad)
__global__ void edge_scatter(const float* __restrict__ exv, const float* __restrict__ den,
                             const float* __restrict__ n_h, const int* __restrict__ src,
                             const int* __restrict__ dst, float* __restrict__ nz) {
  size_t i = (size_t)blockIdx.x * blockDim.x + threadIdx.x;
  if (i >= (size_t)NE * (HD / 4)) return;
  int e = (int)(i >> 5), q = (int)(i & 31);
  int d = dst[e];
  float a = exv[e] / den[d];
  v4f sv = ((const v4f*)(n_h + (size_t)src[e] * HD))[q];
  float* np = nz + (size_t)d * HD + q * 4;
  atomicAdd(np + 0, a * sv[0]);
  atomicAdd(np + 1, a * sv[1]);
  atomicAdd(np + 2, a * sv[2]);
  atomicAdd(np + 3, a * sv[3]);
}

__global__ void add2_v4(const float* __restrict__ a, const float* __restrict__ b,
                        float* __restrict__ o, size_t n4) {
  size_t i = (size_t)blockIdx.x * blockDim.x + threadIdx.x;
  if (i < n4) ((v4f*)o)[i] = ((const v4f*)a)[i] + ((const v4f*)b)[i];
}

__global__ void edge_update(const float* __restrict__ e_h, const float* __restrict__ nz,
                            const int* __restrict__ src, const int* __restrict__ dst,
                            float* __restrict__ eh) {
  size_t i = (size_t)blockIdx.x * blockDim.x + threadIdx.x;
  if (i >= (size_t)NE * (HD / 4)) return;
  int e = (int)(i >> 5), q = (int)(i & 31);
  v4f ev = ((const v4f*)e_h)[i];
  v4f s  = ((const v4f*)(nz + (size_t)src[e] * HD))[q];
  v4f d  = ((const v4f*)(nz + (size_t)dst[e] * HD))[q];
  ((v4f*)eh)[i] = ev * (1.0f + s - d);
}

__global__ void pool_nodes(const float* __restrict__ nh, const int* __restrict__ batch,
                           float* __restrict__ gh) {
  size_t i = (size_t)blockIdx.x * blockDim.x + threadIdx.x;
  if (i >= (size_t)NN * (HD / 4)) return;
  int n = (int)(i >> 5), q = (int)(i & 31);
  v4f v = ((const v4f*)nh)[i];
  float* g = gh + (size_t)batch[n] * HD + q * 4;
  atomicAdd(g + 0, v[0]); atomicAdd(g + 1, v[1]);
  atomicAdd(g + 2, v[2]); atomicAdd(g + 3, v[3]);
}

__global__ void pool_edges(const float* __restrict__ eh, const int* __restrict__ batch,
                           const int* __restrict__ src, float* __restrict__ gh) {
  size_t i = (size_t)blockIdx.x * blockDim.x + threadIdx.x;
  if (i >= (size_t)NE * (HD / 4)) return;
  int e = (int)(i >> 5), q = (int)(i & 31);
  v4f v = ((const v4f*)eh)[i];
  float* g = gh + (size_t)batch[src[e]] * HD + q * 4;
  atomicAdd(g + 0, v[0]); atomicAdd(g + 1, v[1]);
  atomicAdd(g + 2, v[2]); atomicAdd(g + 3, v[3]);
}

// one thread per (row, class, feature-quad)
__global__ void onehot_out(const float* __restrict__ feat, const int* __restrict__ y,
                           float* __restrict__ out, int rows, int ncls) {
  size_t i = (size_t)blockIdx.x * blockDim.x + threadIdx.x;
  size_t total = (size_t)rows * ncls * (FOUT / 4);
  if (i >= total) return;
  int f4 = (int)(i & (FOUT / 4 - 1));
  int c  = (int)((i / (FOUT / 4)) % ncls);
  int r  = (int)(i / ((size_t)(FOUT / 4) * ncls));
  v4f v = {};
  if (y[r] == c) v = ((const v4f*)(feat + (size_t)r * FOUT))[f4];
  ((v4f*)out)[i] = v;
}

// ---------------- host-side orchestration ----------------
struct MLPp { const float *W1, *W2, *b1, *b2; };
struct BNBp { const float *W1, *W2, *b1, *b2, *beta, *gamma; };

static inline int cdiv(long long a, long long b) { return (int)((a + b - 1) / b); }

static void launch_pack(const float* W, __bf16* wp, int K, int Nc, hipStream_t s) {
  int total = ((K + 31) / 32) * (Nc / 16) * 512;
  pack_w<<<cdiv(total, 256), 256, 0, s>>>(W, wp, K, Nc);
}

static void launch_gemm(const float* X, const __bf16* wp, const float* b, float* Y,
                        int M, int K, int Nc, int act, hipStream_t s) {
  bool k32  = (K % 32) == 0;
  int tiles = M / 16;
  int pairs = tiles / 2;
  int rem   = tiles - 2 * pairs;
  dim3 blk(128);
  if (pairs) {
    dim3 grd(cdiv(pairs, 4));
    if (Nc == 128) {
      if (act)      gemm_wmma2<128, 1, true ><<<grd, blk, 0, s>>>(X, wp, b, Y, M, K);
      else if (k32) gemm_wmma2<128, 0, true ><<<grd, blk, 0, s>>>(X, wp, b, Y, M, K);
      else          gemm_wmma2<128, 0, false><<<grd, blk, 0, s>>>(X, wp, b, Y, M, K);
    } else {
      gemm_wmma2<64, 0, true><<<grd, blk, 0, s>>>(X, wp, b, Y, M, K);
    }
  }
  if (rem) {   // one leftover 16-row tile
    const float* Xr = X + (size_t)pairs * 32 * K;
    float*       Yr = Y + (size_t)pairs * 32 * Nc;
    if (Nc == 128) {
      if (act)      gemm_wmma<128, 1, true ><<<1, blk, 0, s>>>(Xr, wp, b, Yr, 16, K);
      else if (k32) gemm_wmma<128, 0, true ><<<1, blk, 0, s>>>(Xr, wp, b, Yr, 16, K);
      else          gemm_wmma<128, 0, false><<<1, blk, 0, s>>>(Xr, wp, b, Yr, 16, K);
    } else {
      gemm_wmma<64, 0, true><<<1, blk, 0, s>>>(Xr, wp, b, Yr, 16, K);
    }
  }
}

static void launch_bn(float* hid, const float* gamma, const float* beta,
                      float* gsum, float* ss, int M, hipStream_t s) {
  hipMemsetAsync(gsum, 0, 2 * HD * sizeof(float), s);
  size_t total4 = (size_t)M * (HD / 4);
  int blocks = cdiv((long long)total4, 256);
  if (blocks > 1024) blocks = 1024;
  col_stats<<<blocks, 256, 0, s>>>(hid, gsum, (size_t)M);
  bn_finalize<<<1, HD, 0, s>>>(gsum, gamma, beta, ss, 1.0f / (float)M);
  bn_gelu<<<cdiv((long long)total4, 256), 256, 0, s>>>(hid, ss, total4);
}

// Linear -> BN(batch stats) -> GELU(exact) -> Linear
static void run_block(const float* X, const BNBp& p, float* Y, float* hid,
                      int M, int K, int NcOut, __bf16* wp, float* gsum, float* ss,
                      hipStream_t s) {
  launch_pack(p.W1, wp, K, HD, s);
  launch_gemm(X, wp, p.b1, hid, M, K, HD, 0, s);
  launch_bn(hid, p.gamma, p.beta, gsum, ss, M, s);
  launch_pack(p.W2, wp, HD, NcOut, s);
  launch_gemm(hid, wp, p.b2, Y, M, HD, NcOut, 0, s);
}

// Linear -> ReLU -> Linear
static void run_mlp(const float* X, const MLPp& p, float* Y, float* hid,
                    int M, __bf16* wp, hipStream_t s) {
  launch_pack(p.W1, wp, HD, HD, s);
  launch_gemm(X, wp, p.b1, hid, M, HD, HD, 1, s);
  launch_pack(p.W2, wp, HD, HD, s);
  launch_gemm(hid, wp, p.b2, Y, M, HD, HD, 0, s);
}

extern "C" void kernel_launch(void* const* d_in, const int* in_sizes, int n_in,
                              void* d_out, int out_size, void* d_ws, size_t ws_size,
                              hipStream_t stream) {
  (void)in_sizes; (void)n_in; (void)out_size; (void)ws_size;
  const float* x  = (const float*)d_in[0];
  const float* ea = (const float*)d_in[1];
  auto F = [&](int i) { return (const float*)d_in[i]; };

  // params flattened in pytree (sorted-key) order:
  // gnn[l]: ef{W1,W2,b1,b2}, nf{W1,W2,b1,b2}; then init_ef, init_nf, out_ef, out_gf, out_nf
  // each BN-mlp: {W1,W2,b1,b2,beta,gamma}
  MLPp gef[3], gnf[3];
  int idx = 2;
  for (int l = 0; l < 3; ++l) {
    gef[l] = { F(idx), F(idx + 1), F(idx + 2), F(idx + 3) }; idx += 4;
    gnf[l] = { F(idx), F(idx + 1), F(idx + 2), F(idx + 3) }; idx += 4;
  }
  BNBp init_ef = { F(idx), F(idx+1), F(idx+2), F(idx+3), F(idx+4), F(idx+5) }; idx += 6;
  BNBp init_nf = { F(idx), F(idx+1), F(idx+2), F(idx+3), F(idx+4), F(idx+5) }; idx += 6;
  BNBp out_ef  = { F(idx), F(idx+1), F(idx+2), F(idx+3), F(idx+4), F(idx+5) }; idx += 6;
  BNBp out_gf  = { F(idx), F(idx+1), F(idx+2), F(idx+3), F(idx+4), F(idx+5) }; idx += 6;
  BNBp out_nf  = { F(idx), F(idx+1), F(idx+2), F(idx+3), F(idx+4), F(idx+5) }; idx += 6;
  const int* edge_index = (const int*)d_in[idx++];
  const int* batch      = (const int*)d_in[idx++];
  const int* node_y     = (const int*)d_in[idx++];
  const int* edge_y     = (const int*)d_in[idx++];
  const int* src = edge_index;
  const int* dst = edge_index + NE;

  // workspace layout (floats, 256B aligned regions)
  float* ws = (float*)d_ws;
  auto al = [](size_t v) { return (v + 63) & ~(size_t)63; };
  size_t o = 0;
  float* nh   = ws + o; o = al(o + (size_t)NN * HD);
  float* eh   = ws + o; o = al(o + (size_t)NE * HD);
  float* n_h  = ws + o; o = al(o + (size_t)NN * HD);
  float* e_h  = ws + o; o = al(o + (size_t)NE * HD);
  float* nz   = ws + o; o = al(o + (size_t)NN * HD);
  float* tmp  = ws + o; o = al(o + (size_t)NE * HD);   // gemm hidden (edge-sized)
  float* attn = ws + o; o = al(o + (size_t)NE);
  float* exv  = ws + o; o = al(o + (size_t)NE);
  float* mval = ws + o; o = al(o + (size_t)NN);
  float* den  = ws + o; o = al(o + (size_t)NN);
  float* ss   = ws + o; o = al(o + 256);
  float* gsum = ws + o; o = al(o + 256);
  float* gh   = ws + o; o = al(o + (size_t)NG * HD);
  float* gh2  = ws + o; o = al(o + (size_t)NG * HD);
  float* nh_o = ws + o; o = al(o + (size_t)NN * FOUT);
  float* eh_o = ws + o; o = al(o + (size_t)NE * FOUT);
  __bf16* wp  = (__bf16*)(ws + o);                      // 16384 bf16 fragment buffer

  // -------- init blocks --------
  run_block(x,  init_nf, nh, tmp, NN, NFIN, HD, wp, gsum, ss, stream);
  run_block(ea, init_ef, eh, tmp, NE, EFIN, HD, wp, gsum, ss, stream);

  // -------- GAT layers --------
  const size_t NNH = (size_t)NN * HD;
  const size_t NE4 = (size_t)NE * (HD / 4), NN4 = (size_t)NN * (HD / 4);
  for (int l = 0; l < 3; ++l) {
    run_mlp(nh, gnf[l], n_h, tmp, NN, wp, stream);
    run_mlp(eh, gef[l], e_h, tmp, NE, wp, stream);
    hipMemsetAsync(nz,  0, NNH * sizeof(float), stream);
    hipMemsetAsync(den, 0, (size_t)NN * sizeof(float), stream);
    fill_f<<<cdiv(NN, 256), 256, 0, stream>>>(mval, -__builtin_inff(), NN);
    edge_attn<<<cdiv(NE, 8), 256, 0, stream>>>(n_h, e_h, src, dst, attn);
    seg_max<<<cdiv(NE, 256), 256, 0, stream>>>(attn, dst, mval);
    edge_exp<<<cdiv(NE, 256), 256, 0, stream>>>(attn, mval, dst, exv, den);
    edge_scatter<<<cdiv((long long)NE4, 256), 256, 0, stream>>>(exv, den, n_h, src, dst, nz);
    add2_v4<<<cdiv((long long)NN4, 256), 256, 0, stream>>>(n_h, nz, nh, NN4);
    edge_update<<<cdiv((long long)NE4, 256), 256, 0, stream>>>(e_h, nz, src, dst, eh);
  }

  // -------- pooling + graph head (writes straight into d_out) --------
  hipMemsetAsync(gh, 0, (size_t)NG * HD * sizeof(float), stream);
  pool_nodes<<<cdiv((long long)NN4, 256), 256, 0, stream>>>(nh, batch, gh);
  pool_edges<<<cdiv((long long)NE4, 256), 256, 0, stream>>>(eh, batch, src, gh);
  float* out    = (float*)d_out;
  float* gh_out = out + (size_t)NN * (NCLS * FOUT) + (size_t)NE * (ECLS * FOUT);
  run_block(gh, out_gf, gh_out, gh2, NG, HD, HD, wp, gsum, ss, stream);

  // -------- node / edge heads + one-hot expansion --------
  run_block(nh, out_nf, nh_o, tmp, NN, HD, FOUT, wp, gsum, ss, stream);
  run_block(eh, out_ef, eh_o, tmp, NE, HD, FOUT, wp, gsum, ss, stream);
  onehot_out<<<cdiv((long long)NN * NCLS * (FOUT / 4), 256), 256, 0, stream>>>(
      nh_o, node_y, out, NN, NCLS);
  onehot_out<<<cdiv((long long)NE * ECLS * (FOUT / 4), 256), 256, 0, stream>>>(
      eh_o, edge_y, out + (size_t)NN * (NCLS * FOUT), NE, ECLS);
}
